// UncollapsedGibbsIBP_55714315764213
// MI455X (gfx1250) — matrix operand order repaired and checked by reading, not compiled, not run on hardware
//
#include <hip/hip_runtime.h>
#include <math.h>

// Problem dims (fixed by reference)
#define N_ 1024
#define K_ 32
#define D_ 256
#define T_ 1024
#define LD_ (D_ + K_ + T_ + N_)   // 2336 output row stride

#define SIGMA2_A 1.0f
#define SIGMA2_N 0.1f
#define PHI      0.3f
#define EPS_     0.05f
#define LAMBD    0.8f
#define LOGIT_PHI (-0.84729786038720340f)   // log(PHI/(1-PHI))

typedef __attribute__((ext_vector_type(16))) _Float16     v16h;
typedef __attribute__((ext_vector_type(8)))  _Float16     h8;
typedef __attribute__((ext_vector_type(8)))  float        v8f;
typedef __attribute__((ext_vector_type(4)))  unsigned int u32x4;
typedef __attribute__((ext_vector_type(8)))  unsigned int u32x8;

// ---------------- workspace layout (bytes) ----------------
#define PLANE    ((size_t)N_ * T_ * 2)          // one f16 likelihood plane (2MB)
#define LL0_OFF  ((size_t)0)                    // g_0   row-major (N x T)
#define LLM_OFF  (LL0_OFF + PLANE)              // g_-1
#define LLP_OFF  (LLM_OFF + PLANE)              // g_+1
#define LT0_OFF  (LLP_OFF + PLANE)              // g_0^T  (T x N)
#define LTM_OFF  (LT0_OFF + PLANE)
#define LTP_OFF  (LTM_OFF + PLANE)
#define ZH_OFF   (LTP_OFF + PLANE)              // Z  f16 (N x K)
#define ZTH_OFF  (ZH_OFF  + (size_t)N_*K_*2)    // Z^T f16 (K x N)
#define YH_OFF   (ZTH_OFF + (size_t)N_*K_*2)    // Y  f16 (K x T)
#define YTH_OFF  (YH_OFF  + (size_t)K_*T_*2)    // Y^T f16 (T x K)
#define F32_OFF  (YTH_OFF + (size_t)K_*T_*2)
// f32 region offsets (in floats, from F32_OFF)
#define O_TAB0   0
#define O_TAB1   64
#define O_ZCS    128
#define O_G      160
#define O_GINV   (O_G    + K_*K_)
#define O_ZTF    (O_GINV + K_*K_)
#define O_AAT    (O_ZTF  + K_*D_)
#define O_FAT    (O_AAT  + K_*K_)
#define O_M      (O_FAT  + N_*K_)       // 3 * K*T
#define O_P      (O_M    + 3*K_*T_)     // 3 * N*K

__device__ __forceinline__ float sigmoidf_(float x) { return 1.0f / (1.0f + __expf(-x)); }

__device__ __forceinline__ v16h cat16(h8 lo, h8 hi) {
    return __builtin_shufflevector(lo, hi, 0,1,2,3,4,5,6,7,8,9,10,11,12,13,14,15);
}

// ---------------------------------------------------------------------------
// setup: 34-entry log tables for noisy-OR, Z column sums
__global__ void k_setup(const float* __restrict__ Z, float* __restrict__ f32) {
    int tid = threadIdx.x;
    if (tid < 34) {
        float q = powf(1.0f - LAMBD, (float)tid) * (1.0f - EPS_);
        f32[O_TAB0 + tid] = logf(1.0f - q);   // log p1(n)
        f32[O_TAB1 + tid] = logf(q);          // log (1-p1(n))
    }
    if (tid >= 64 && tid < 96) {
        int k = tid - 64; float s = 0.f;
        for (int i = 0; i < N_; ++i) s += Z[i*K_ + k];
        f32[O_ZCS + k] = s;
    }
}

// ---------------------------------------------------------------------------
// f16 operand copies in the orientations the WMMA GEMMs want
__global__ void k_half(const float* __restrict__ Z, const float* __restrict__ Y,
                       _Float16* __restrict__ zh,  _Float16* __restrict__ zth,
                       _Float16* __restrict__ yh,  _Float16* __restrict__ yth) {
    int idx = blockIdx.x * blockDim.x + threadIdx.x;
    if (idx < N_*K_) {
        int i = idx / K_, k = idx % K_;
        _Float16 v = (_Float16)Z[idx];
        zh[idx] = v;  zth[k*N_ + i] = v;
    } else if (idx < N_*K_ + K_*T_) {
        int j = idx - N_*K_;
        int k = j / T_, t = j % T_;
        _Float16 v = (_Float16)Y[j];
        yh[j] = v;  yth[t*K_ + k] = v;
    }
}

// ---------------------------------------------------------------------------
// G = Z^T Z + ridge I ; ZtF = Z^T F
__global__ void k_gztf(const float* __restrict__ Z, const float* __restrict__ F,
                       float* __restrict__ f32) {
    int idx = blockIdx.x * blockDim.x + threadIdx.x;
    if (idx < K_*K_) {
        int k1 = idx / K_, k2 = idx % K_;
        float s = 0.f;
        for (int i = 0; i < N_; ++i) s += Z[i*K_ + k1] * Z[i*K_ + k2];
        f32[O_G + idx] = s + ((k1 == k2) ? (SIGMA2_N / SIGMA2_A) : 0.f);
    } else if (idx < K_*K_ + K_*D_) {
        int j = idx - K_*K_;
        int k = j / D_, d = j % D_;
        float s = 0.f;
        for (int i = 0; i < N_; ++i) s += Z[i*K_ + k] * F[i*D_ + d];
        f32[O_ZTF + j] = s;
    }
}

// ---------------------------------------------------------------------------
// 32x32 Gauss-Jordan inverse (SPD + ridge -> no pivoting). Writes covA.
__global__ void k_inv(float* __restrict__ f32, float* __restrict__ out) {
    __shared__ float Mx[32][65];
    __shared__ float fac[32];
    int tid = threadIdx.x;
    for (int idx = tid; idx < 32*64; idx += 256) {
        int r = idx >> 6, c = idx & 63;
        Mx[r][c] = (c < 32) ? f32[O_G + r*32 + c] : ((c - 32 == r) ? 1.f : 0.f);
    }
    __syncthreads();
    for (int p = 0; p < 32; ++p) {
        float pv = Mx[p][p];
        __syncthreads();
        if (tid < 64) Mx[p][tid] /= pv;
        if (tid < 32) fac[tid] = (tid == p) ? 0.f : Mx[tid][p];
        __syncthreads();
        for (int idx = tid; idx < 32*64; idx += 256) {
            int r = idx >> 6, c = idx & 63;
            if (r != p) Mx[r][c] -= fac[r] * Mx[p][c];
        }
        __syncthreads();
    }
    for (int idx = tid; idx < 32*32; idx += 256) {
        int r = idx >> 5, c = idx & 31;
        float v = Mx[r][32 + c];
        f32[O_GINV + idx] = v;
        out[r*LD_ + D_ + c] = SIGMA2_N * v;   // covA
    }
}

// ---------------------------------------------------------------------------
// AAt = A A^T ; FAt = F A^T   (f32 VALU; small)
__global__ void k_aat_fat(const float* __restrict__ F, const float* __restrict__ A,
                          float* __restrict__ f32) {
    int idx = blockIdx.x * blockDim.x + threadIdx.x;
    if (idx < K_*K_) {
        int k1 = idx / K_, k2 = idx % K_;
        float s = 0.f;
        for (int d = 0; d < D_; ++d) s += A[k1*D_ + d] * A[k2*D_ + d];
        f32[O_AAT + idx] = s;
    } else if (idx < K_*K_ + N_*K_) {
        int j = idx - K_*K_;
        int i = j / K_, k = j % K_;
        float s = 0.f;
        for (int d = 0; d < D_; ++d) s += F[i*D_ + d] * A[k*D_ + d];
        f32[O_FAT + j] = s;
    }
}

// muA = Ginv @ ZtF
__global__ void k_mua(const float* __restrict__ f32, float* __restrict__ out) {
    int idx = blockIdx.x * blockDim.x + threadIdx.x;  // K*D
    int k = idx / D_, d = idx % D_;
    float s = 0.f;
    for (int j = 0; j < K_; ++j) s += f32[O_GINV + k*K_ + j] * f32[O_ZTF + j*D_ + d];
    out[k*LD_ + d] = s;
}

// ---------------------------------------------------------------------------
// WMMA fragment layouts (ISA 7.12.2, wave32):
//   A 16x32 f16: lane l -> row M=l%16 ; half e -> kk = 16*(e/8) + 8*(l/16) + (e%8)
//   B 32x16 f16: lane l -> col N=l%16 ; half e -> kk = 16*(e/8) + 8*(l/16) + (e%8)
//   C 16x16 f32: vgpr r, lane l -> M = r + 8*(l/16), N = l%16
// With reduction-contiguous layouts each fragment is two aligned 16B loads.

// ---------------------------------------------------------------------------
// prep: baseY tile via one v_wmma_f32_16x16x32_f16 (K=32, exact in f16 since
// Z,Y binary), then LDS table lookup -> g_s planes in BOTH orientations.
__global__ void k_prep(const _Float16* __restrict__ zh, const _Float16* __restrict__ yth,
                       const float* __restrict__ X, const float* __restrict__ f32,
                       _Float16* __restrict__ g0,  _Float16* __restrict__ gm,
                       _Float16* __restrict__ gp,  _Float16* __restrict__ g0t,
                       _Float16* __restrict__ gmt, _Float16* __restrict__ gpt) {
    __shared__ float s0[34], s1[34];
    if (threadIdx.x < 34) {
        s0[threadIdx.x] = f32[O_TAB0 + threadIdx.x];
        s1[threadIdx.x] = f32[O_TAB1 + threadIdx.x];
    }
    __syncthreads();

    int wave = (blockIdx.x * blockDim.x + threadIdx.x) >> 5;  // 0..4095
    int lane = threadIdx.x & 31;
    int i0 = (wave >> 6) << 4;
    int t0 = (wave & 63) << 4;
    int mrow = lane & 15, hi = lane >> 4;

    // A = Z rows (reduction k contiguous in zh)
    h8 alo = *(const h8*)&zh[(i0 + mrow)*K_ + hi*8];
    h8 ahi = *(const h8*)&zh[(i0 + mrow)*K_ + 16 + hi*8];
    // B = Y (column-major via yth: reduction k contiguous)
    h8 blo = *(const h8*)&yth[(t0 + mrow)*K_ + hi*8];
    h8 bhi = *(const h8*)&yth[(t0 + mrow)*K_ + 16 + hi*8];
    v16h a = cat16(alo, ahi), b = cat16(blo, bhi);
    v8f c = {0.f,0.f,0.f,0.f,0.f,0.f,0.f,0.f};
    c = __builtin_amdgcn_wmma_f32_16x16x32_f16(false, a, false, b, (short)0, c, false, false);

    for (int r = 0; r < 8; ++r) {
        int i = i0 + r + (hi << 3);
        int t = t0 + mrow;
        int n = (int)(c[r] + 0.5f);               // integer count in [0,32]
        const float* tab = (X[i*T_ + t] > 0.5f) ? s0 : s1;
        int nm = (n > 0) ? (n - 1) : 0;           // s=-1 never used at n==0
        _Float16 v0 = (_Float16)tab[n];
        _Float16 vm = (_Float16)tab[nm];
        _Float16 vp = (_Float16)tab[n + 1];
        g0 [i*T_ + t] = v0;  gm [i*T_ + t] = vm;  gp [i*T_ + t] = vp;
        g0t[t*N_ + i] = v0;  gmt[t*N_ + i] = vm;  gpt[t*N_ + i] = vp;
    }
}

// ---------------------------------------------------------------------------
// M_s = Z^T @ g_s (K x T). Block = 8 waves covering a 32x64 slab of one plane.
// Per 32-step of the N reduction:
//   - Z^T chunk (32k x 32i, 2KB) staged via per-lane global_load_async_to_lds_b128
//   - g^T chunk (64t x 32i, 4KB) staged via one TDM tensor_load_to_lds (wave 0)
// Fragments then read from LDS as aligned b128s (reduction contiguous).
__global__ void k_M(const _Float16* __restrict__ zth,
                    const _Float16* __restrict__ g0t, const _Float16* __restrict__ gmt,
                    const _Float16* __restrict__ gpt, float* __restrict__ f32) {
    __shared__ _Float16 smZ[32*32];   // [k][i]
    __shared__ _Float16 smG[64*32];   // [t][i]
    int mat = blockIdx.x >> 4;                 // 0..2
    int t0b = (blockIdx.x & 15) << 6;          // 64-col slab
    const _Float16* gT = (mat == 0) ? g0t : (mat == 1) ? gmt : gpt;

    int tid = threadIdx.x, lane = tid & 31, wv = tid >> 5;
    int k0  = (wv >> 2) << 4;                  // wave k-tile: 0 or 16
    int t0l = (wv & 3) << 4;                   // wave t-tile within slab
    int mrow = lane & 15, hi = lane >> 4;

    unsigned int ldsZ = (unsigned int)(size_t)(void*)smZ;
    unsigned int ldsG = (unsigned int)(size_t)(void*)smG;

    v8f c = {0.f,0.f,0.f,0.f,0.f,0.f,0.f,0.f};
    for (int ii = 0; ii < N_; ii += 32) {
        __syncthreads();   // previous iteration fully consumed
        // ---- Z^T chunk: threads 0..127 copy one 16B piece each (async path)
        if (tid < 128) {
            int r = tid >> 2, ch = tid & 3;                       // 32 rows x 4 chunks
            unsigned int lofs = ldsZ + (unsigned int)(r*64 + ch*16);
            unsigned int gofs = (unsigned int)((r*N_ + ch*8) * 2); // bytes from (zth+ii)
            asm volatile("global_load_async_to_lds_b128 %0, %1, %2"
                         :: "v"(lofs), "v"(gofs), "s"(zth + ii) : "memory");
        }
        // ---- g^T chunk: one Tensor-Data-Mover descriptor, issued by wave 0
        if (wv == 0) {
            unsigned long long ga = (unsigned long long)(size_t)(gT + (size_t)t0b*N_ + ii);
            u32x4 d0; u32x8 d1;
            d0[0] = 1u;                                    // count=1, no gather
            d0[1] = ldsG;                                  // lds_addr
            d0[2] = (unsigned int)ga;                      // global_addr[31:0]
            d0[3] = ((unsigned int)(ga >> 32) & 0x01FFFFFFu) | 0x80000000u; // [56:32], type=2
            d1[0] = (1u << 16);                            // wg_mask=0, data_size=2B
            d1[1] = ((unsigned int)(N_ & 0xFFFF)) << 16;   // tensor_dim0[15:0]
            d1[2] = ((unsigned int)(N_ >> 16) & 0xFFFFu) | (((unsigned int)(T_ & 0xFFFF)) << 16); // dim0 hi | dim1 lo
            d1[3] = ((unsigned int)(T_ >> 16) & 0xFFFFu) | (32u << 16);    // dim1 hi | tile_dim0=32
            d1[4] = 64u;                                   // tile_dim1=64, tile_dim2=0
            d1[5] = (unsigned int)N_;                      // tensor_dim0_stride
            d1[6] = 0u;                                    // stride0 hi, stride1 lo
            d1[7] = 0u;                                    // stride1 hi
            asm volatile("tensor_load_to_lds %0, %1" :: "s"(d0), "s"(d1) : "memory");
        }
        if (tid < 128) asm volatile("s_wait_asynccnt 0" ::: "memory");
        if (wv == 0) __builtin_amdgcn_s_wait_tensorcnt(0);
        __syncthreads();

        // ---- fragments from LDS (all aligned b128 reads)
        h8 alo = *(const h8*)&smZ[(k0 + mrow)*32 + hi*8];
        h8 ahi = *(const h8*)&smZ[(k0 + mrow)*32 + 16 + hi*8];
        h8 blo = *(const h8*)&smG[(t0l + mrow)*32 + hi*8];
        h8 bhi = *(const h8*)&smG[(t0l + mrow)*32 + 16 + hi*8];
        v16h a = cat16(alo, ahi), b = cat16(blo, bhi);
        c = __builtin_amdgcn_wmma_f32_16x16x32_f16(false, a, false, b, (short)0, c, false, false);
    }
    float* Mo = f32 + O_M + (size_t)mat * K_ * T_;
    for (int r = 0; r < 8; ++r)
        Mo[(k0 + r + (hi << 3))*T_ + t0b + t0l + mrow] = c[r];
}

// ---------------------------------------------------------------------------
// P_s = g_s @ Y^T (N x K). Both operands reduction-contiguous in global
// (g row-major, Yh row-major) -> pure b128 streams, prefetch hints.
__global__ void k_P(const _Float16* __restrict__ yh,
                    const _Float16* __restrict__ g0, const _Float16* __restrict__ gm,
                    const _Float16* __restrict__ gp, float* __restrict__ f32) {
    int wave = (blockIdx.x * blockDim.x + threadIdx.x) >> 5;  // 0..383
    int lane = threadIdx.x & 31;
    int mat  = wave >> 7;
    int rem  = wave & 127;
    int i0   = (rem >> 1) << 4;
    int k0   = (rem & 1) << 4;
    const _Float16* g = (mat == 0) ? g0 : (mat == 1) ? gm : gp;
    int mrow = lane & 15, hi = lane >> 4;

    v8f c = {0.f,0.f,0.f,0.f,0.f,0.f,0.f,0.f};
    for (int tt = 0; tt < T_; tt += 32) {
        if (tt + 32 < T_) {
            __builtin_prefetch(&g [(i0 + mrow)*T_ + tt + 32], 0, 1);
            __builtin_prefetch(&yh[(k0 + mrow)*T_ + tt + 32], 0, 1);
        }
        h8 alo = *(const h8*)&g [(i0 + mrow)*T_ + tt + hi*8];
        h8 ahi = *(const h8*)&g [(i0 + mrow)*T_ + tt + 16 + hi*8];
        h8 blo = *(const h8*)&yh[(k0 + mrow)*T_ + tt + hi*8];
        h8 bhi = *(const h8*)&yh[(k0 + mrow)*T_ + tt + 16 + hi*8];
        v16h a = cat16(alo, ahi), b = cat16(blo, bhi);
        c = __builtin_amdgcn_wmma_f32_16x16x32_f16(false, a, false, b, (short)0, c, false, false);
    }
    float* Po = f32 + O_P + (size_t)mat * N_ * K_;
    for (int r = 0; r < 8; ++r)
        Po[(i0 + r + (hi << 3))*K_ + k0 + mrow] = c[r];
}

// ---------------------------------------------------------------------------
// pY1[k,t] = sigmoid(logit(phi) + (Y ? M0-Mm1 : Mp1-M0))
__global__ void k_finalY(const float* __restrict__ Y, const float* __restrict__ f32,
                         float* __restrict__ out) {
    int idx = blockIdx.x * blockDim.x + threadIdx.x;   // K*T == k*T + t
    int k = idx >> 10, t = idx & (T_ - 1);
    const float* M0 = f32 + O_M;
    const float* Mm = M0 + K_*T_;
    const float* Mp = Mm + K_*T_;
    float d = (Y[idx] > 0.5f) ? (M0[idx] - Mm[idx]) : (Mp[idx] - M0[idx]);
    out[k*LD_ + D_ + K_ + t] = sigmoidf_((float)LOGIT_PHI + d);
}

// ---------------------------------------------------------------------------
// pZ1[i,k]: prior log-odds + noisy-OR delta + Gaussian delta.
__global__ void k_finalZ(const float* __restrict__ Z, const float* __restrict__ f32,
                         float* __restrict__ out) {
    int idx = blockIdx.x * blockDim.x + threadIdx.x;   // N*K == i*K + k
    int i = idx >> 5, k = idx & (K_ - 1);
    float z = Z[idx];
    float m = f32[O_ZCS + k] - z;
    float p = 0.f;
    if (m > 0.f) {
        float ms  = fminf(fmaxf(m, 1e-6f), (float)N_ - 1e-6f);
        float dpr = logf(ms) - logf((float)N_ - ms);
        const float* P0 = f32 + O_P;
        const float* Pm = P0 + N_*K_;
        const float* Pp = Pm + N_*K_;
        float dx = (z > 0.5f) ? (P0[idx] - Pm[idx]) : (Pp[idx] - P0[idx]);
        float ra = f32[O_FAT + idx];
        for (int j = 0; j < K_; ++j) ra -= Z[i*K_ + j] * f32[O_AAT + j*K_ + k];
        float asq = f32[O_AAT + k*K_ + k];
        float df  = (ra - (0.5f - z) * asq) * (1.0f / SIGMA2_N);
        p = sigmoidf_(dpr + dx + df);
    }
    out[k*LD_ + D_ + K_ + T_ + i] = p;
}

// ---------------------------------------------------------------------------
extern "C" void kernel_launch(void* const* d_in, const int* in_sizes, int n_in,
                              void* d_out, int out_size, void* d_ws, size_t ws_size,
                              hipStream_t stream) {
    (void)in_sizes; (void)n_in; (void)out_size; (void)ws_size;
    const float* F = (const float*)d_in[0];
    const float* X = (const float*)d_in[1];
    const float* Z = (const float*)d_in[2];
    const float* Y = (const float*)d_in[3];
    const float* A = (const float*)d_in[4];
    float*       out = (float*)d_out;

    char* ws = (char*)d_ws;
    _Float16* g0  = (_Float16*)(ws + LL0_OFF);
    _Float16* gm  = (_Float16*)(ws + LLM_OFF);
    _Float16* gp  = (_Float16*)(ws + LLP_OFF);
    _Float16* g0t = (_Float16*)(ws + LT0_OFF);
    _Float16* gmt = (_Float16*)(ws + LTM_OFF);
    _Float16* gpt = (_Float16*)(ws + LTP_OFF);
    _Float16* zh  = (_Float16*)(ws + ZH_OFF);
    _Float16* zth = (_Float16*)(ws + ZTH_OFF);
    _Float16* yh  = (_Float16*)(ws + YH_OFF);
    _Float16* yth = (_Float16*)(ws + YTH_OFF);
    float*    f32 = (float*)(ws + F32_OFF);

    k_setup  <<<1,   128, 0, stream>>>(Z, f32);
    k_half   <<<256, 256, 0, stream>>>(Z, Y, zh, zth, yh, yth);
    k_gztf   <<<36,  256, 0, stream>>>(Z, F, f32);
    k_inv    <<<1,   256, 0, stream>>>(f32, out);
    k_aat_fat<<<132, 256, 0, stream>>>(F, A, f32);
    k_mua    <<<32,  256, 0, stream>>>(f32, out);
    k_prep   <<<512, 256, 0, stream>>>(zh, yth, X, f32, g0, gm, gp, g0t, gmt, gpt);
    k_M      <<<48,  256, 0, stream>>>(zth, g0t, gmt, gpt, f32);
    k_P      <<<48,  256, 0, stream>>>(yh, g0, gm, gp, f32);
    k_finalY <<<128, 256, 0, stream>>>(Y, f32, out);
    k_finalZ <<<128, 256, 0, stream>>>(Z, f32, out);
}
